// MultiHeadAttention_32177894981640
// MI455X (gfx1250) — matrix operand order repaired and checked by reading, not compiled, not run on hardware
//
#include <hip/hip_runtime.h>
#include <hip/hip_bf16.h>
#include <math.h>

typedef __attribute__((ext_vector_type(16))) _Float16 v16h;
typedef __attribute__((ext_vector_type(8)))  float    v8f;
typedef __attribute__((ext_vector_type(4)))  float    f32x4;   // trivial 16B vector
typedef __attribute__((ext_vector_type(4)))  int      i32x4;   // 16B int vector (builtin arg type)

union frag16 { v16h v; f32x4 f[2]; };   // one WMMA f16 operand = 32B/lane

#define BATCH 4
#define SEQ   2048
#define EMBD  768
#define NHEAD 12
#define HD    64
#define SCALE_F 0.03608439182435161f   /* 1/sqrt(768) per reference */

static __device__ __forceinline__ v8f wmma16x16x32(const v16h& a, const v16h& b, const v8f& c) {
  return __builtin_amdgcn_wmma_f32_16x16x32_f16(false, a, false, b, (short)0, c, false, false);
}

// ---- async global->LDS (CDNA5 GLOBAL_LOAD_ASYNC_TO_LDS_B128, ASYNCcnt) ----
#if __has_builtin(__builtin_amdgcn_global_load_async_to_lds_b128) && \
    __has_builtin(__builtin_amdgcn_s_wait_asynccnt)
#define HAVE_ASYNC_LDS 1
#else
#define HAVE_ASYNC_LDS 0
#endif

__device__ __forceinline__ void copy16_to_lds(const _Float16* g, _Float16* l) {
#if HAVE_ASYNC_LDS
  // builtin expects 16B-vector pointers (v4i*): cast via AS-void* -> AS-v4i*
  __attribute__((address_space(1))) void* gv =
      (__attribute__((address_space(1))) void*)(void*)g;
  __attribute__((address_space(3))) void* lv =
      (__attribute__((address_space(3))) void*)(void*)l;
  __builtin_amdgcn_global_load_async_to_lds_b128(
      (__attribute__((address_space(1))) i32x4*)gv,
      (__attribute__((address_space(3))) i32x4*)lv, 0, 0);
#else
  *(f32x4*)l = *(const f32x4*)g;
#endif
}
__device__ __forceinline__ void async_wait_all() {
#if HAVE_ASYNC_LDS
  __builtin_amdgcn_s_wait_asynccnt(0);
#endif
}

// lane-xor shuffle within 16-lane groups via ds_swizzle (no index VGPR needed)
#if __has_builtin(__builtin_amdgcn_ds_swizzle)
#define XOR_F(v, m) __int_as_float(__builtin_amdgcn_ds_swizzle(__float_as_int(v), (0x1f | ((m) << 10))))
#else
#define XOR_F(v, m) __shfl_xor((v), (m), 32)
#endif
__device__ __forceinline__ float red_max16(float v) {
  v = fmaxf(v, XOR_F(v, 1)); v = fmaxf(v, XOR_F(v, 2));
  v = fmaxf(v, XOR_F(v, 4)); v = fmaxf(v, XOR_F(v, 8));
  return v;
}
__device__ __forceinline__ float red_sum16(float v) {
  v += XOR_F(v, 1); v += XOR_F(v, 2); v += XOR_F(v, 4); v += XOR_F(v, 8);
  return v;
}

// ---------------------------------------------------------------------------
__global__ void cvt_f32_to_f16(const float* __restrict__ src, _Float16* __restrict__ dst, int n) {
  int i = blockIdx.x * blockDim.x + threadIdx.x;
  int stride = gridDim.x * blockDim.x;
  for (; i < n; i += stride) dst[i] = (_Float16)src[i];
}

// ---------------------------------------------------------------------------
// Tiled f16 GEMM: C[M,N] = A[M,K]*B[K,N] + bias. 256 thr = 8 waves (4x2),
// tile 128x128, BK=32, double-buffered LDS. A staged by async-to-LDS; B staged
// column-major (manual transpose) so WMMA fragments are two ds_load_b128.
// MODE 0: QKV scatter epilogue (f16 Q*scale / K / V). MODE 1: f32 out.
// ---------------------------------------------------------------------------
#define TSTR 40   // padded LDS stride (halves); 80B, 16B aligned
template <int MODE>
__global__ __launch_bounds__(256) void gemm_f16(
    const _Float16* __restrict__ A, const _Float16* __restrict__ Bm,
    const float* __restrict__ bias,
    _Float16* __restrict__ Qh, _Float16* __restrict__ Kh, _Float16* __restrict__ Vh,
    float* __restrict__ outF, int M, int N, int K)
{
  __shared__ __align__(16) _Float16 sA[2][128 * TSTR];   // row-major  [m][k]
  __shared__ __align__(16) _Float16 sB[2][128 * TSTR];   // col-major  [n][k]

  const int tid   = threadIdx.x;
  const int lane  = tid & 31;
  const int wave  = tid >> 5;
  const int hl    = lane >> 4;
  const int l16   = lane & 15;
  const int waveM = wave >> 1;
  const int waveN = wave & 1;
  const int bm = blockIdx.y * 128;
  const int bn = blockIdx.x * 128;

  auto stage = [&](int buf, int k0) {
    // A tile 128x32: 512 16B chunks -> async straight into LDS
    for (int q = tid; q < 512; q += 256) {
      int r = q >> 2, cs = q & 3;
      copy16_to_lds(&A[(size_t)(bm + r) * K + k0 + cs * 8],
                    &sA[buf][r * TSTR + cs * 8]);
    }
    // B tile 32x128 -> LDS column-major (k innermost per column)
    for (int q = tid; q < 512; q += 256) {
      int r = q >> 4, cs = q & 15;
      f32x4 tmp = *(const f32x4*)(&Bm[(size_t)(k0 + r) * N + bn + cs * 8]);
      const _Float16* tp = (const _Float16*)&tmp;
#pragma unroll
      for (int j = 0; j < 8; ++j) sB[buf][(cs * 8 + j) * TSTR + r] = tp[j];
    }
  };

  v8f acc[2][4];
#pragma unroll
  for (int tm = 0; tm < 2; ++tm)
#pragma unroll
    for (int tn = 0; tn < 4; ++tn) acc[tm][tn] = v8f{};

  stage(0, 0);
  int cur = 0;
  for (int k0 = 0; k0 < K; k0 += 32) {
    async_wait_all();
    __syncthreads();
    if (k0 + 32 < K) {
      stage(cur ^ 1, k0 + 32);
      __builtin_prefetch(&A[(size_t)(bm + (tid >> 1)) * K + k0 + 64], 0, 1);
      __builtin_prefetch(&Bm[(size_t)(k0 + 64 + (tid >> 4)) * N + bn], 0, 1);
    }
    frag16 aF[2], bF[4];
#pragma unroll
    for (int tm = 0; tm < 2; ++tm) {
      const f32x4* p = (const f32x4*)&sA[cur][(waveM * 32 + tm * 16 + l16) * TSTR];
      aF[tm].f[0] = p[hl];          // K = hl*8 + 0..7
      aF[tm].f[1] = p[2 + hl];      // K = 16 + hl*8 + 0..7
    }
#pragma unroll
    for (int tn = 0; tn < 4; ++tn) {
      const f32x4* p = (const f32x4*)&sB[cur][(waveN * 64 + tn * 16 + l16) * TSTR];
      bF[tn].f[0] = p[hl];          // K = hl*8 + 0..7
      bF[tn].f[1] = p[2 + hl];      // K = 16 + hl*8 + 0..7
    }
#pragma unroll
    for (int tm = 0; tm < 2; ++tm)
#pragma unroll
      for (int tn = 0; tn < 4; ++tn)
        acc[tm][tn] = wmma16x16x32(aF[tm].v, bF[tn].v, acc[tm][tn]);
    cur ^= 1;
  }

#pragma unroll
  for (int tm = 0; tm < 2; ++tm)
#pragma unroll
    for (int tn = 0; tn < 4; ++tn)
#pragma unroll
      for (int r = 0; r < 8; ++r) {
        int row = bm + waveM * 32 + tm * 16 + r + hl * 8;
        int col = bn + waveN * 64 + tn * 16 + l16;
        float val = acc[tm][tn][r] + bias[col];
        if (MODE == 0) {
          int h = col / 192, rem = col % 192;
          int d = rem / 3, s = rem % 3;
          int bI = row >> 11, nI = row & 2047;
          size_t idx = (((size_t)bI * NHEAD + h) * SEQ + nI) * HD + d;
          if (s == 0)      Qh[idx] = (_Float16)(val * SCALE_F);
          else if (s == 1) Kh[idx] = (_Float16)val;
          else             Vh[idx] = (_Float16)val;
        } else {
          outF[(size_t)row * N + col] = val;
        }
      }
}

// ---------------------------------------------------------------------------
// Flash attention. Grid (SEQ/128, BATCH*NHEAD), 8 waves; wave = 16 query rows.
// Q and K staged via async-to-LDS (row-major: QK^T B-fragments contiguous);
// V staged transposed (sVt[d][key]) so PV B-fragments are contiguous.
// Double-buffered K/V; P region overlays the (dead after init) Q tile.
// ---------------------------------------------------------------------------
#define KVSTR 72   // 72 halves = 144B, 16B aligned
__global__ __launch_bounds__(256) void flash_attn(
    const _Float16* __restrict__ Qh, const _Float16* __restrict__ Kh,
    const _Float16* __restrict__ Vh, _Float16* __restrict__ Oh)
{
  __shared__ __align__(16) _Float16 smem[9216 + 2 * 64 * KVSTR + 2 * 64 * KVSTR];
  _Float16* sQ = smem;
  _Float16* sP = smem;                       // overlays sQ (sQ dead after init)
  _Float16* sKb = smem + 9216;
  _Float16* sVb = smem + 9216 + 2 * 64 * KVSTR;

  const int tid  = threadIdx.x;
  const int lane = tid & 31;
  const int wave = tid >> 5;
  const int hl   = lane >> 4;
  const int l16  = lane & 15;
  const int qb = blockIdx.x * 128;
  const int bh = blockIdx.y;
  const int bI = bh / NHEAD, h = bh % NHEAD;

  const _Float16* Qg = Qh + ((size_t)bh * SEQ + qb) * HD;
  const _Float16* Kg = Kh + (size_t)bh * SEQ * HD;
  const _Float16* Vg = Vh + (size_t)bh * SEQ * HD;

  // ---- stage Q tile 128x64 (async) and build persistent A-fragments
  for (int q = tid; q < 1024; q += 256) {
    int r = q >> 3, cs = q & 7;
    copy16_to_lds(&Qg[(size_t)r * HD + cs * 8], &sQ[r * KVSTR + cs * 8]);
  }
  async_wait_all();
  __syncthreads();
  frag16 aQ[2];
  {
    const f32x4* p = (const f32x4*)&sQ[(wave * 16 + l16) * KVSTR];
#pragma unroll
    for (int c = 0; c < 2; ++c) {
      aQ[c].f[0] = p[c * 4 + hl];
      aQ[c].f[1] = p[c * 4 + 2 + hl];
    }
  }

  auto stageKV = [&](int buf, int kb) {
    const _Float16* Kt = Kg + (size_t)kb * 64 * HD;
    const _Float16* Vt = Vg + (size_t)kb * 64 * HD;
    _Float16* dK = sKb + buf * 64 * KVSTR;
    _Float16* dV = sVb + buf * 64 * KVSTR;
    for (int q = tid; q < 512; q += 256) {       // K row-major [key][d], async
      int key = q >> 3, ds = q & 7;
      copy16_to_lds(&Kt[(size_t)key * HD + ds * 8], &dK[key * KVSTR + ds * 8]);
    }
    for (int q = tid; q < 512; q += 256) {       // V transposed -> [d][key]
      int key = q >> 3, ds = q & 7;
      f32x4 tmp = *(const f32x4*)(&Vt[(size_t)key * HD + ds * 8]);
      const _Float16* tp = (const _Float16*)&tmp;
#pragma unroll
      for (int j = 0; j < 8; ++j) dV[(ds * 8 + j) * KVSTR + key] = tp[j];
    }
  };

  float mrow[8], lrow[8];
  v8f oacc[4];
#pragma unroll
  for (int i = 0; i < 8; ++i) { mrow[i] = -__builtin_inff(); lrow[i] = 0.f; }
#pragma unroll
  for (int t = 0; t < 4; ++t) oacc[t] = v8f{};

  _Float16* sPw = sP + wave * 16 * 64;

  stageKV(0, 0);
  int cur = 0;
  for (int kb = 0; kb < SEQ / 64; ++kb) {
    async_wait_all();
    __syncthreads();
    if (kb + 1 < SEQ / 64) {
      stageKV(cur ^ 1, kb + 1);
      __builtin_prefetch(Kg + (size_t)(kb + 2) * 64 * HD + (tid << 3), 0, 1);
      __builtin_prefetch(Vg + (size_t)(kb + 2) * 64 * HD + (tid << 3), 0, 1);
    }
    const _Float16* cK = sKb + cur * 64 * KVSTR;
    const _Float16* cV = sVb + cur * 64 * KVSTR;

    // ---- S = Q*K^T : 4 col tiles x 2 K-chunks; B frags contiguous from cK
    v8f sacc[4];
#pragma unroll
    for (int j = 0; j < 4; ++j) sacc[j] = v8f{};
#pragma unroll
    for (int j = 0; j < 4; ++j) {
      const f32x4* p = (const f32x4*)&cK[(j * 16 + l16) * KVSTR];
#pragma unroll
      for (int c = 0; c < 2; ++c) {
        frag16 bK;
        bK.f[0] = p[c * 4 + hl];
        bK.f[1] = p[c * 4 + 2 + hl];
        sacc[j] = wmma16x16x32(aQ[c].v, bK.v, sacc[j]);
      }
    }

    // ---- online softmax
    float fscale[8];
#pragma unroll
    for (int i = 0; i < 8; ++i) {
      float tmax = sacc[0][i];
#pragma unroll
      for (int j = 1; j < 4; ++j) tmax = fmaxf(tmax, sacc[j][i]);
      tmax = red_max16(tmax);
      float mn = fmaxf(mrow[i], tmax);
      float f  = __expf(mrow[i] - mn);
      float ts = 0.f;
#pragma unroll
      for (int j = 0; j < 4; ++j) {
        float pql = __expf(sacc[j][i] - mn);
        sacc[j][i] = pql;
        ts += pql;
      }
      ts = red_sum16(ts);
      lrow[i] = lrow[i] * f + ts;
      mrow[i] = mn;
      fscale[i] = f;
    }
#pragma unroll
    for (int t = 0; t < 4; ++t)
#pragma unroll
      for (int i = 0; i < 8; ++i) oacc[t][i] *= fscale[i];

    // ---- P (C layout) -> wave-private LDS, reload as A fragments
#pragma unroll
    for (int j = 0; j < 4; ++j)
#pragma unroll
      for (int i = 0; i < 8; ++i)
        sPw[(i + hl * 8) * 64 + j * 16 + l16] = (_Float16)sacc[j][i];

    // ---- O += P*V
#pragma unroll
    for (int c = 0; c < 2; ++c) {
      frag16 aP;
      const f32x4* pp = (const f32x4*)&sPw[l16 * 64];
      aP.f[0] = pp[c * 4 + hl];
      aP.f[1] = pp[c * 4 + 2 + hl];
#pragma unroll
      for (int t = 0; t < 4; ++t) {
        frag16 bV;
        const f32x4* pv = (const f32x4*)&cV[(t * 16 + l16) * KVSTR];
        bV.f[0] = pv[c * 4 + hl];
        bV.f[1] = pv[c * 4 + 2 + hl];
        oacc[t] = wmma16x16x32(aP.v, bV.v, oacc[t]);
      }
    }
    cur ^= 1;
  }

  // ---- normalize, write O as [b, n, h*d]
#pragma unroll
  for (int t = 0; t < 4; ++t)
#pragma unroll
    for (int i = 0; i < 8; ++i) {
      int row = qb + wave * 16 + i + hl * 8;
      float val = oacc[t][i] / lrow[i];
      Oh[((size_t)bI * SEQ + row) * EMBD + h * HD + t * 16 + l16] = (_Float16)val;
    }
}

// ---------------------------------------------------------------------------
extern "C" void kernel_launch(void* const* d_in, const int* in_sizes, int n_in,
                              void* d_out, int out_size, void* d_ws, size_t ws_size,
                              hipStream_t stream) {
  (void)in_sizes; (void)n_in; (void)out_size; (void)ws_size;
  const float* x     = (const float*)d_in[0];
  const float* Wqkv  = (const float*)d_in[1];
  const float* bqkv  = (const float*)d_in[2];
  const float* Wproj = (const float*)d_in[3];
  const float* bproj = (const float*)d_in[4];
  float* out = (float*)d_out;

  const int M = BATCH * SEQ;
  char* ws = (char*)d_ws;
  size_t off = 0;
  auto take = [&](size_t nelem) {
    _Float16* p = (_Float16*)(ws + off);
    off += ((nelem * sizeof(_Float16) + 255) / 256) * 256;
    return p;
  };
  _Float16* xh     = take((size_t)M * EMBD);
  _Float16* wqkvh  = take((size_t)EMBD * 3 * EMBD);
  _Float16* wprojh = take((size_t)EMBD * EMBD);
  _Float16* Qh     = take((size_t)BATCH * NHEAD * SEQ * HD);
  _Float16* Kh     = take((size_t)BATCH * NHEAD * SEQ * HD);
  _Float16* Vh     = take((size_t)BATCH * NHEAD * SEQ * HD);
  _Float16* Oh     = take((size_t)M * EMBD);

  cvt_f32_to_f16<<<2048, 256, 0, stream>>>(x, xh, M * EMBD);
  cvt_f32_to_f16<<<1024, 256, 0, stream>>>(Wqkv, wqkvh, EMBD * 3 * EMBD);
  cvt_f32_to_f16<<<512,  256, 0, stream>>>(Wproj, wprojh, EMBD * EMBD);

  gemm_f16<0><<<dim3(3 * EMBD / 128, M / 128), 256, 0, stream>>>(
      xh, wqkvh, bqkv, Qh, Kh, Vh, nullptr, M, 3 * EMBD, EMBD);

  flash_attn<<<dim3(SEQ / 128, BATCH * NHEAD), 256, 0, stream>>>(Qh, Kh, Vh, Oh);

  gemm_f16<1><<<dim3(EMBD / 128, M / 128), 256, 0, stream>>>(
      Oh, wprojh, bproj, nullptr, nullptr, nullptr, out, M, EMBD, EMBD);
}
